// CQT_fit_11725260718516
// MI455X (gfx1250) — matrix operand order repaired and checked by reading, not compiled, not run on hardware
//
#include <hip/hip_runtime.h>
#include <hip/hip_bf16.h>

// ---- problem constants (batch/layout fixed by the reference) ----
#define BATCH   32
#define NBINS   84
#define ROWS    168          // NBINS * 2 (real/imag)
#define ROWSP   176          // padded to 11 tiles of 16
#define MTILES  11
#define HOPSZ   512
#define WAVES   4            // waves per workgroup (wave32)
#define BLOCK   (WAVES * 32)
#define KSTEP   64           // taps per TDM transfer / barrier (2 WMMA K-chunks)
#define LDS_STRIDE 72        // halves per LDS row (144B = 36 dwords -> conflict-free b128)
#define LDS_BUF  (ROWSP * LDS_STRIDE)   // halves per buffer (25344 B)

typedef __attribute__((ext_vector_type(16))) _Float16 v16h;
typedef __attribute__((ext_vector_type(8)))  _Float16 v8h;
typedef __attribute__((ext_vector_type(8)))  float    v8f;
typedef __attribute__((ext_vector_type(4)))  unsigned int u32x4;
typedef __attribute__((ext_vector_type(4)))  int      i32x4;
typedef __attribute__((ext_vector_type(8)))  int      i32x8;

// ---------------- pre-pass: fp32 -> f16 with zero padding ----------------

__global__ void cqt_cvt_x(const float* __restrict__ x, _Float16* __restrict__ xh,
                          int T, int pad, long xstride) {
    long i = (long)blockIdx.x * blockDim.x + threadIdx.x;
    if (i >= xstride) return;
    int b = blockIdx.y;
    long s = i - pad;
    float v = (s >= 0 && s < T) ? x[(size_t)b * T + s] : 0.0f;
    xh[(size_t)b * xstride + i] = (_Float16)v;
}

__global__ void cqt_cvt_k(const float* __restrict__ K, _Float16* __restrict__ kh,
                          int Lmax, int kpad) {
    int k = blockIdx.x * blockDim.x + threadIdx.x;
    int r = blockIdx.y;
    if (k >= kpad) return;
    float v = (r < ROWS && k < Lmax) ? K[(size_t)r * Lmax + k] : 0.0f;
    kh[(size_t)r * kpad + k] = (_Float16)v;
}

// ---------------- TDM: DMA one 176x64-half kernel slice into LDS ----------------
//
// 2-D tensor descriptor, data_size=2B. LDS padding: after every 32 DWORDs
// (= 64 halves = one tile row) insert 4 DWORDs (= 8 halves) -> LDS row
// stride of 72 halves, keeping ds_load_b128 A-fragment reads conflict-free.

__device__ __forceinline__ void tdm_load_tile(unsigned int lds_off,
                                              const _Float16* gsrc, int kpad) {
    unsigned long long ga = (unsigned long long)(uintptr_t)gsrc;
    u32x4 g0;
    g0[0] = 1u;                                   // count=1, user mode, no gather
    g0[1] = lds_off;                              // lds_addr (bytes, WG-relative)
    g0[2] = (unsigned int)ga;                     // global_addr[31:0]
    g0[3] = (unsigned int)((ga >> 32) & 0x01FFFFFFu) | (2u << 30); // addr[56:32] | type=2

    unsigned int kp = (unsigned int)kpad;
    i32x8 g1;
    g1[0] = (int)((1u << 16)      // data_size = 2 bytes
                | (1u << 20)      // pad_enable
                | (4u << 22)      // pad_interval: 32 DWORDs (128B = one row)
                | (3u << 25));    // pad_amount: 4 DWORDs (16B)
    g1[1] = (int)((kp & 0xFFFFu) << 16);          // tensor_dim0[15:0] (no atomic barrier)
    g1[2] = (int)((kp >> 16) | ((unsigned)ROWSP << 16)); // tensor_dim0 hi | tensor_dim1 lo
    g1[3] = (int)((unsigned)KSTEP << 16);         // tensor_dim1 hi=0 | tile_dim0=64
    g1[4] = (int)ROWSP;                           // tile_dim1=176 | tile_dim2=0
    g1[5] = (int)kp;                              // tensor_dim0_stride[31:0] (elements)
    g1[6] = 0;                                    // stride hi | dim1_stride lo (unused, 2D)
    g1[7] = 0;

    i32x4 gz = {0, 0, 0, 0};
#if defined(__clang_major__) && (__clang_major__ >= 23)
    i32x8 gz8 = {0, 0, 0, 0, 0, 0, 0, 0};
    __builtin_amdgcn_tensor_load_to_lds(g0, g1, gz, gz, gz8, 0);
#else
    __builtin_amdgcn_tensor_load_to_lds(g0, g1, gz, gz, 0);
#endif
}

// ---------------- main kernel: WMMA f16 GEMM over im2col'd x ----------------

__global__ __launch_bounds__(BLOCK) void cqt_wmma(
    const _Float16* __restrict__ xh, const _Float16* __restrict__ kh,
    float* __restrict__ out, int T_out, int kpad, long xstride) {

    const int b    = blockIdx.y;
    const int tid  = threadIdx.x;
    const int lane = tid & 31;
    const int wave = tid >> 5;
    const int t0   = (blockIdx.x * WAVES + wave) * 16;

    __shared__ __align__(16) _Float16 sK[2][LDS_BUF];
    const unsigned int lds0 = (unsigned int)(uintptr_t)(void*)&sK[0][0];
    const unsigned int lds1 = (unsigned int)(uintptr_t)(void*)&sK[1][0];

    v8f acc[MTILES];
    #pragma unroll
    for (int i = 0; i < MTILES; ++i)
        #pragma unroll
        for (int j = 0; j < 8; ++j) acc[i][j] = 0.0f;

    // B fragment: lane holds column N = lane&15 (time t), K range [kbB,kbB+16)
    const int tcol = t0 + (lane & 15);
    const int kbB  = (lane < 16) ? 0 : 16;
    const _Float16* xwin = xh + (size_t)b * xstride + (size_t)tcol * HOPSZ + kbB;

    // A fragment (LDS): lane holds row M = lane&15, halves [0..15]=K kbA..kbA+15
    // (relative to the 32-wide K-chunk)
    const int kbA = (lane < 16) ? 0 : 16;
    const int mA  = lane & 15;

    const int nk = kpad / KSTEP;   // K-steps of 64

    // prologue: DMA first kernel slice into buffer 0; preload B(0) into regs
    if (wave == 0) tdm_load_tile(lds0, kh, kpad);

    v8h b0lo = *(const v8h*)(xwin);
    v8h b0hi = *(const v8h*)(xwin + 8);
    v8h b1lo = *(const v8h*)(xwin + 32);
    v8h b1hi = *(const v8h*)(xwin + 40);
    v16h bfrag0 = __builtin_shufflevector(b0lo, b0hi,
        0,1,2,3,4,5,6,7,8,9,10,11,12,13,14,15);
    v16h bfrag1 = __builtin_shufflevector(b1lo, b1hi,
        0,1,2,3,4,5,6,7,8,9,10,11,12,13,14,15);

    for (int kt = 0; kt < nk; ++kt) {
        const int cur = kt & 1;
        if (wave == 0) __builtin_amdgcn_s_wait_tensorcnt(0);  // slice kt landed
        __syncthreads();  // publish buf[cur]; all waves done reading buf[cur^1]
        if (wave == 0 && (kt + 1) < nk)                        // overlap DMA kt+1
            tdm_load_tile(cur ? lds0 : lds1, kh + (size_t)(kt + 1) * KSTEP, kpad);

        // issue B(kt+1) now; consumed next iteration (hidden by 22 WMMAs).
        const int k0n = (kt + 1 < nk) ? ((kt + 1) * KSTEP) : (kt * KSTEP);
        v8h n0lo = *(const v8h*)(xwin + k0n);
        v8h n0hi = *(const v8h*)(xwin + k0n + 8);
        v8h n1lo = *(const v8h*)(xwin + k0n + 32);
        v8h n1hi = *(const v8h*)(xwin + k0n + 40);
        __builtin_prefetch(xwin + k0n + KSTEP, 0, 0);

        const _Float16* sbuf = &sK[cur][0];

        // software-pipelined 22-stage loop: stage s -> (kk = s<11?0:32, mt = s%11)
        const _Float16* p0 = sbuf + mA * LDS_STRIDE + kbA;
        v8h alo = *(const v8h*)(p0);
        v8h ahi = *(const v8h*)(p0 + 8);
        #pragma unroll
        for (int s = 0; s < 2 * MTILES; ++s) {
            const int mt = (s < MTILES) ? s : (s - MTILES);
            v16h afrag = __builtin_shufflevector(alo, ahi,
                0,1,2,3,4,5,6,7,8,9,10,11,12,13,14,15);
            if (s + 1 < 2 * MTILES) {
                const int sn  = s + 1;
                const int mtn = (sn < MTILES) ? sn : (sn - MTILES);
                const int kkn = (sn < MTILES) ? 0 : 32;
                const _Float16* pa = sbuf + (mtn * 16 + mA) * LDS_STRIDE + kkn + kbA;
                alo = *(const v8h*)(pa);
                ahi = *(const v8h*)(pa + 8);
            }
            acc[mt] = __builtin_amdgcn_wmma_f32_16x16x32_f16(
                false, afrag, false, (s < MTILES) ? bfrag0 : bfrag1,
                (short)0, acc[mt], false, false);
        }

        bfrag0 = __builtin_shufflevector(n0lo, n0hi,
            0,1,2,3,4,5,6,7,8,9,10,11,12,13,14,15);
        bfrag1 = __builtin_shufflevector(n1lo, n1hi,
            0,1,2,3,4,5,6,7,8,9,10,11,12,13,14,15);
    }

    // C/D layout: lanes 0-15 -> N=lane, M=v; lanes 16-31 -> N=lane-16, M=v+8
    if (tcol < T_out) {
        const int mbase = (lane < 16) ? 0 : 8;
        #pragma unroll
        for (int mt = 0; mt < MTILES; ++mt) {
            #pragma unroll
            for (int v = 0; v < 8; ++v) {
                int r = mt * 16 + mbase + v;
                if (r < ROWS) {
                    int bin = r >> 1, c = r & 1;
                    out[(((size_t)b * 2 + c) * NBINS + bin) * T_out + tcol] = acc[mt][v];
                }
            }
        }
    }
}

// ---------------- launch ----------------

extern "C" void kernel_launch(void* const* d_in, const int* in_sizes, int n_in,
                              void* d_out, int out_size, void* d_ws, size_t ws_size,
                              hipStream_t stream) {
    const float* x = (const float*)d_in[0];     // (32, T) fp32
    const float* K = (const float*)d_in[1];     // (84, 2, Lmax) fp32
    float* out = (float*)d_out;                 // (32, 2, 84, T_out) fp32

    const int T     = in_sizes[0] / BATCH;
    const int Lmax  = in_sizes[1] / ROWS;
    const int T_out = out_size / (BATCH * ROWS);
    const int pad   = Lmax / 2;
    const int kpad  = ((Lmax + KSTEP - 1) / KSTEP) * KSTEP;

    const int ntile = (T_out + 15) / 16;
    const int nwg   = (ntile + WAVES - 1) / WAVES;
    long xstride = ((long)(nwg * WAVES * 16 - 1) * HOPSZ + kpad + 15) & ~15L;

    _Float16* xh = (_Float16*)d_ws;
    _Float16* kh = xh + (size_t)BATCH * xstride;

    cqt_cvt_x<<<dim3((unsigned)((xstride + 255) / 256), BATCH), 256, 0, stream>>>(
        x, xh, T, pad, xstride);
    cqt_cvt_k<<<dim3((kpad + 255) / 256, ROWSP), 256, 0, stream>>>(
        K, kh, Lmax, kpad);
    cqt_wmma<<<dim3(nwg, BATCH), BLOCK, 0, stream>>>(
        xh, kh, out, T_out, kpad, xstride);
}
